// SpatioTemporalGAT_76321568850117
// MI455X (gfx1250) — matrix operand order repaired and checked by reading, not compiled, not run on hardware
//
#include <hip/hip_runtime.h>

// ---------------- problem constants (match reference) ----------------
#define N_NODES   100000                 // divisible by 16 -> GEMM tiles are always full
#define N_EDGES   1600000
#define ET        (N_EDGES + N_NODES)   // edges + self loops
#define IN_F      32
#define D_E       8
#define HID       64
#define HEADS     4
#define CH        16
#define NEG_SLOPE 0.2f

static_assert(N_NODES % 16 == 0, "GEMM assumes full 16-row tiles");

typedef __attribute__((ext_vector_type(16))) _Float16 v16h;
typedef __attribute__((ext_vector_type(8)))  float    v8f;

// Monotonic float <-> uint encoding so segment-max can use u32 atomicMax.
__device__ __forceinline__ unsigned fenc(float x) {
    unsigned u = __float_as_uint(x);
    return (u & 0x80000000u) ? ~u : (u | 0x80000000u);
}
__device__ __forceinline__ float fdec(unsigned e) {
    unsigned u = (e & 0x80000000u) ? (e ^ 0x80000000u) : ~e;
    return __uint_as_float(u);
}

// ---------------- generic u32 fill (also used to zero f32 buffers) ----------------
__global__ void fill_u32(unsigned* __restrict__ p, unsigned v, long n) {
    long i = (long)blockIdx.x * blockDim.x + threadIdx.x;
    if (i < n) p[i] = v;
}

// ---------------- fold We (D_E x 64) and a_edge (HEADS x CH) into q[h][d] ----------------
__global__ void edge_q_kernel(const float* __restrict__ We, const float* __restrict__ ae,
                              float* __restrict__ q) {
    int gid = threadIdx.x;                 // 32 threads: h*8 + d
    if (gid >= HEADS * D_E) return;
    int h = gid >> 3, d = gid & 7;
    float s = 0.f;
    #pragma unroll
    for (int c = 0; c < CH; ++c)
        s += We[(size_t)d * HID + h * CH + c] * ae[h * CH + c];
    q[gid] = s;
}

// ---------------- WMMA GEMM: Y[nrows,64] = X[nrows,K] @ W[K,64]  (f16 in, f32 acc) ----
// One wave per FULL 16-row tile; 4 column tiles of 16. nrows must be a multiple of 16,
// so there are no partial tiles: loads are unclamped, stores are unguarded (EXEC stays
// all-ones for the entire kernel, as WMMA requires).
template <int K>
__global__ void gemm_wmma(const float* __restrict__ X, const float* __restrict__ W,
                          float* __restrict__ Y, int nrows) {
    const int lane  = threadIdx.x & 31;
    const int wave  = threadIdx.x >> 5;
    const int gw    = blockIdx.x * (blockDim.x >> 5) + wave;
    const int row0  = gw * 16;
    if (row0 + 16 > nrows) return;         // wave-uniform exit; no partial tiles exist

    const int rhalf = lane >> 4;           // 0: lanes 0-15, 1: lanes 16-31
    const int rl    = lane & 15;
    const int r     = row0 + rl;

    // A per ISA 16-bit A layout: lanes<16: elems 0-7 = K 0-7, 8-15 = K 16-23;
    //                            lanes>=16: elems 0-7 = K 8-15, 8-15 = K 24-31.
    v16h a[K / 32];
    #pragma unroll
    for (int kc = 0; kc < K / 32; ++kc) {
        const float* xp = X + (size_t)r * K + kc * 32 + rhalf * 8;
        #pragma unroll
        for (int i = 0; i < 8; ++i) {
            a[kc][i]     = (_Float16)xp[i];
            a[kc][8 + i] = (_Float16)xp[16 + i];
        }
    }

    #pragma unroll
    for (int ct = 0; ct < 4; ++ct) {
        v8f c = {};
        const int col = ct * 16 + rl;
        #pragma unroll
        for (int kc = 0; kc < K / 32; ++kc) {
            // B same K order as A: lanes<16 hold K 0-7,16-23; lanes>=16 hold K 8-15,24-31.
            v16h b;
            const int kb = kc * 32 + rhalf * 8;
            #pragma unroll
            for (int i = 0; i < 16; ++i)
                b[i] = (_Float16)W[(size_t)(kb + (i < 8 ? i : i + 8)) * HID + col];
            c = __builtin_amdgcn_wmma_f32_16x16x32_f16(
                    false, a[kc], false, b, (short)0, c, false, false);
        }
        // C/D layout: VGPR j -> row = row0 + j + 8*rhalf, col = ct*16 + rl.
        // Unconditional stores: straight-line epilogue, no exec-mask churn.
        float* yp = Y + (size_t)(row0 + rhalf * 8) * HID + col;
        #pragma unroll
        for (int j = 0; j < 8; ++j)
            yp[(size_t)j * HID] = c[j];
    }
}

// ---------------- per-node attention scores: sn/dn[i,h] = <h[i,h,:], a_{src/dst}[h,:]> --
__global__ void node_scores(const float* __restrict__ H, const float* __restrict__ a_s,
                            const float* __restrict__ a_d, float* __restrict__ sn,
                            float* __restrict__ dn, int n) {
    int gid = blockIdx.x * blockDim.x + threadIdx.x;
    if (gid >= n * HEADS) return;
    int i = gid >> 2, h = gid & 3;
    const float* hp = H + (size_t)i * HID + h * CH;
    float s = 0.f, d = 0.f;
    #pragma unroll
    for (int c = 0; c < CH; ++c) {
        float v = hp[c];
        s += v * a_s[h * CH + c];
        d += v * a_d[h * CH + c];
    }
    sn[gid] = s;
    dn[gid] = d;
}

// ---------------- alpha = leakyrelu(sn[src]+dn[dst]+<eattr,q>), segment max -------------
__global__ void alpha_kernel(const int* __restrict__ ei, const float* __restrict__ eattr,
                             const float* __restrict__ sla, const float* __restrict__ q,
                             const float* __restrict__ sn, const float* __restrict__ dn,
                             float* __restrict__ alpha, unsigned* __restrict__ mx) {
    long t = (long)blockIdx.x * blockDim.x + threadIdx.x;
    if (t >= ET) return;
    int s, dd;
    float ea[D_E];
    if (t < N_EDGES) {
        s  = ei[t];
        dd = ei[(long)N_EDGES + t];
        #pragma unroll
        for (int d = 0; d < D_E; ++d) ea[d] = eattr[t * D_E + d];
    } else {
        s = dd = (int)(t - N_EDGES);
        #pragma unroll
        for (int d = 0; d < D_E; ++d) ea[d] = sla[d];
    }
    #pragma unroll
    for (int h = 0; h < HEADS; ++h) {
        float se = 0.f;
        #pragma unroll
        for (int d = 0; d < D_E; ++d) se += ea[d] * q[h * D_E + d];
        float al = sn[(size_t)s * HEADS + h] + dn[(size_t)dd * HEADS + h] + se;
        al = (al > 0.f) ? al : NEG_SLOPE * al;
        alpha[t * HEADS + h] = al;
        atomicMax(&mx[(size_t)dd * HEADS + h], fenc(al));
    }
}

// ---------------- expa = exp(alpha - mx[dst]); den = segment_sum(expa) ------------------
__global__ void expsum_kernel(const int* __restrict__ ei, float* __restrict__ alpha,
                              const unsigned* __restrict__ mx, float* __restrict__ den) {
    long t = (long)blockIdx.x * blockDim.x + threadIdx.x;
    if (t >= ET) return;
    int dd = (t < N_EDGES) ? ei[(long)N_EDGES + t] : (int)(t - N_EDGES);
    #pragma unroll
    for (int h = 0; h < HEADS; ++h) {
        float m = fdec(mx[(size_t)dd * HEADS + h]);
        float e = __expf(alpha[t * HEADS + h] - m);
        alpha[t * HEADS + h] = e;
        atomicAdd(&den[(size_t)dd * HEADS + h], e);
    }
}

// ---------------- acc[dst] += h[src] * coef   (one thread per edge x head) --------------
__global__ void aggregate_kernel(const int* __restrict__ ei, const float* __restrict__ H,
                                 const float* __restrict__ alpha, const float* __restrict__ den,
                                 float* __restrict__ acc) {
    long gid = (long)blockIdx.x * blockDim.x + threadIdx.x;
    if (gid >= (long)ET * HEADS) return;
    long t = gid >> 2;
    int  h = (int)(gid & 3);
    int s, dd;
    if (t < N_EDGES) {
        s  = ei[t];
        dd = ei[(long)N_EDGES + t];
    } else {
        s = dd = (int)(t - N_EDGES);
    }
    float coef = alpha[t * HEADS + h] / (den[(size_t)dd * HEADS + h] + 1e-16f);
    const float* hp = H   + (size_t)s  * HID + h * CH;
    float*       op = acc + (size_t)dd * HID + h * CH;
    #pragma unroll
    for (int c = 0; c < CH; ++c) atomicAdd(&op[c], hp[c] * coef);
}

// ---------------- out = relu(acc + bias) in place ---------------------------------------
__global__ void bias_relu(float* __restrict__ acc, const float* __restrict__ bias, int n) {
    int gid = blockIdx.x * blockDim.x + threadIdx.x;
    if (gid >= n * HID) return;
    float v = acc[gid] + bias[gid & (HID - 1)];
    acc[gid] = v > 0.f ? v : 0.f;
}

// ---------------- head MLP: out[i] = lw2 . relu(H[i]@lw1 + lb1) + lb2 -------------------
__global__ void mlp_kernel(const float* __restrict__ H, const float* __restrict__ lw1,
                           const float* __restrict__ lb1, const float* __restrict__ lw2,
                           const float* __restrict__ lb2, float* __restrict__ out, int n) {
    int i = blockIdx.x * blockDim.x + threadIdx.x;
    if (i >= n) return;
    float hid[8];
    #pragma unroll
    for (int j = 0; j < 8; ++j) hid[j] = lb1[j];
    const float* hp = H + (size_t)i * HID;
    for (int c = 0; c < HID; ++c) {
        float v = hp[c];
        #pragma unroll
        for (int j = 0; j < 8; ++j) hid[j] += v * lw1[c * 8 + j];
    }
    float o = lb2[0];
    #pragma unroll
    for (int j = 0; j < 8; ++j) {
        float r = hid[j] > 0.f ? hid[j] : 0.f;
        o += r * lw2[j];
    }
    out[i] = o;
}

// ============================= host side =============================
static inline int nblk(long n, int tb) { return (int)((n + tb - 1) / tb); }

extern "C" void kernel_launch(void* const* d_in, const int* in_sizes, int n_in,
                              void* d_out, int out_size, void* d_ws, size_t ws_size,
                              hipStream_t stream) {
    const float* x     = (const float*)d_in[0];
    const int*   ei    = (const int*)d_in[1];     // [2,E]
    const float* eattr = (const float*)d_in[2];   // [E,8]
    const float* W1    = (const float*)d_in[3];
    const float* We1   = (const float*)d_in[4];
    const float* as1   = (const float*)d_in[5];
    const float* ad1   = (const float*)d_in[6];
    const float* ae1   = (const float*)d_in[7];
    const float* b1    = (const float*)d_in[8];
    const float* W2    = (const float*)d_in[9];
    const float* We2   = (const float*)d_in[10];
    const float* as2   = (const float*)d_in[11];
    const float* ad2   = (const float*)d_in[12];
    const float* ae2   = (const float*)d_in[13];
    const float* b2    = (const float*)d_in[14];
    const float* lw1   = (const float*)d_in[15];
    const float* lb1   = (const float*)d_in[16];
    const float* lw2   = (const float*)d_in[17];
    const float* lb2   = (const float*)d_in[18];
    const float* sla   = (const float*)d_in[19];
    float* out = (float*)d_out;

    // workspace carve-up (floats)
    float*    bufA  = (float*)d_ws;                       // [N,64] projection h
    float*    bufB  = bufA + (size_t)N_NODES * HID;       // [N,64] aggregate / layer out
    float*    sn    = bufB + (size_t)N_NODES * HID;       // [N,4]
    float*    dn    = sn   + (size_t)N_NODES * HEADS;     // [N,4]
    unsigned* mx    = (unsigned*)(dn + (size_t)N_NODES * HEADS);  // [N,4] encoded max
    float*    den   = (float*)(mx + (size_t)N_NODES * HEADS);     // [N,4]
    float*    alpha = den + (size_t)N_NODES * HEADS;      // [Et,4]
    float*    q     = alpha + (size_t)ET * HEADS;         // [4,8]

    const int TB = 256;
    const unsigned ENC_NEG_INF = 0x007FFFFFu;             // fenc(-inf)
    const int gemm_grid = nblk((N_NODES + 15) / 16, TB / 32);

    // =================== layer 1 ===================
    edge_q_kernel<<<1, 32, 0, stream>>>(We1, ae1, q);
    gemm_wmma<IN_F><<<gemm_grid, TB, 0, stream>>>(x, W1, bufA, N_NODES);
    node_scores<<<nblk((long)N_NODES * HEADS, TB), TB, 0, stream>>>(bufA, as1, ad1, sn, dn, N_NODES);
    fill_u32<<<nblk((long)N_NODES * HEADS, TB), TB, 0, stream>>>(mx, ENC_NEG_INF, (long)N_NODES * HEADS);
    fill_u32<<<nblk((long)N_NODES * HEADS, TB), TB, 0, stream>>>((unsigned*)den, 0u, (long)N_NODES * HEADS);
    fill_u32<<<nblk((long)N_NODES * HID, TB), TB, 0, stream>>>((unsigned*)bufB, 0u, (long)N_NODES * HID);
    alpha_kernel<<<nblk(ET, TB), TB, 0, stream>>>(ei, eattr, sla, q, sn, dn, alpha, mx);
    expsum_kernel<<<nblk(ET, TB), TB, 0, stream>>>(ei, alpha, mx, den);
    aggregate_kernel<<<nblk((long)ET * HEADS, TB), TB, 0, stream>>>(ei, bufA, alpha, den, bufB);
    bias_relu<<<nblk((long)N_NODES * HID, TB), TB, 0, stream>>>(bufB, b1, N_NODES);

    // =================== layer 2 ===================
    edge_q_kernel<<<1, 32, 0, stream>>>(We2, ae2, q);
    gemm_wmma<HID><<<gemm_grid, TB, 0, stream>>>(bufB, W2, bufA, N_NODES);
    node_scores<<<nblk((long)N_NODES * HEADS, TB), TB, 0, stream>>>(bufA, as2, ad2, sn, dn, N_NODES);
    fill_u32<<<nblk((long)N_NODES * HEADS, TB), TB, 0, stream>>>(mx, ENC_NEG_INF, (long)N_NODES * HEADS);
    fill_u32<<<nblk((long)N_NODES * HEADS, TB), TB, 0, stream>>>((unsigned*)den, 0u, (long)N_NODES * HEADS);
    fill_u32<<<nblk((long)N_NODES * HID, TB), TB, 0, stream>>>((unsigned*)bufB, 0u, (long)N_NODES * HID);
    alpha_kernel<<<nblk(ET, TB), TB, 0, stream>>>(ei, eattr, sla, q, sn, dn, alpha, mx);
    expsum_kernel<<<nblk(ET, TB), TB, 0, stream>>>(ei, alpha, mx, den);
    aggregate_kernel<<<nblk((long)ET * HEADS, TB), TB, 0, stream>>>(ei, bufA, alpha, den, bufB);
    bias_relu<<<nblk((long)N_NODES * HID, TB), TB, 0, stream>>>(bufB, b2, N_NODES);

    // =================== head MLP ===================
    mlp_kernel<<<nblk(N_NODES, TB), TB, 0, stream>>>(bufB, lw1, lb1, lw2, lb2, out, N_NODES);
}
